// LSTM__17557826306135
// MI455X (gfx1250) — matrix operand order repaired and checked by reading, not compiled, not run on hardware
//
#include <hip/hip_runtime.h>
#include <hip/hip_bf16.h>

#define B_ 128
#define T_ 512
#define I_ 256
#define H_ 512

typedef __attribute__((ext_vector_type(16))) __bf16 v16bf;
typedef __attribute__((ext_vector_type(8)))  float  v8f;

__device__ __forceinline__ unsigned short f2bf(float f) {
    unsigned u = __float_as_uint(f);
    unsigned r = (u + 0x7FFFu + ((u >> 16) & 1u)) >> 16;   // RNE
    return (unsigned short)r;
}

union AFrag { v16bf v; uint4 q[2]; };

// A-matrix fragment (16x32 bf16), ISA layout:
// lanes 0-15 : row M=lane,     K = k0+{0..7}  in v0..3, K = k0+{16..23} in v4..7
// lanes 16-31: row M=lane-16,  K = k0+{8..15},           K = k0+{24..31}
__device__ __forceinline__ v16bf load_a(const unsigned short* rowp, int k, int lane) {
    const unsigned short* p = rowp + k + ((lane >> 4) << 3);
    AFrag u;
    u.q[0] = *(const uint4*)p;
    u.q[1] = *(const uint4*)(p + 16);
    return u.v;
}

// B-matrix fragment (32x16 bf16), B[k][n] = W[n][k] (weights are [N,K] row-major):
// lane holds N = lane&15, K = k0+8*(lane>>4)+{0..7} and k0+16+8*(lane>>4)+{0..7} (same as A)
__device__ __forceinline__ v16bf load_b(const unsigned short* rowp, int k, int lane) {
    const unsigned short* p = rowp + k + ((lane >> 4) << 3);
    AFrag u;
    u.q[0] = *(const uint4*)p;
    u.q[1] = *(const uint4*)(p + 16);
    return u.v;
}

__device__ __forceinline__ v8f wmma_bf16(v16bf a, v16bf b, v8f c) {
    return __builtin_amdgcn_wmma_f32_16x16x32_bf16(false, a, false, b, (short)0, c,
                                                   false, false);
}

__device__ __forceinline__ float sigm(float x) { return 1.0f / (1.0f + __expf(-x)); }

// ---------------------------------------------------------------- conversions
__global__ void cvt_f32_bf16(const float* __restrict__ s, unsigned short* __restrict__ d, int n) {
    int i = blockIdx.x * blockDim.x + threadIdx.x;
    int stride = gridDim.x * blockDim.x;
    for (; i < n; i += stride) d[i] = f2bf(s[i]);
}

// ---------------------------------------------------------------- step kernel A:
// z_{i,f,c} = x_t@Wx^T + h@Wh^T (+ c@Wcf^T for f); c_new = f*c + i*tanh(zc)
struct GateParams {
    const unsigned short *x, *Wii, *Whi, *Wif, *Whf, *Wcf, *Wic, *Whc;
    const float *bi, *bfv, *bc;
    const unsigned short *h_in, *c_in;
    unsigned short* c_outb;      // bf16 c_new (other parity buffer)
    float* c_f32;                // fp32 cell state (lives in d_out)
    int t;
};

__global__ __launch_bounds__(256) void lstm_gates_ifc(GateParams P) {
    const int lane = threadIdx.x & 31;
    const int wave = threadIdx.x >> 5;
    const int lm   = lane & 15;
    const int m0   = blockIdx.y * 64 + (wave & 3) * 16;
    const int n0   = blockIdx.x * 64 + (wave >> 2) * 32;
    const int n1   = n0 + 16;

    const unsigned short* xr = P.x + (size_t)(m0 + lm) * (T_ * I_) + (size_t)P.t * I_;
    const unsigned short* hr = P.h_in + (m0 + lm) * H_;
    const unsigned short* cr = P.c_in + (m0 + lm) * H_;

    const unsigned short* wii0 = P.Wii + (n0 + lm) * I_;
    const unsigned short* wii1 = P.Wii + (n1 + lm) * I_;
    const unsigned short* wif0 = P.Wif + (n0 + lm) * I_;
    const unsigned short* wif1 = P.Wif + (n1 + lm) * I_;
    const unsigned short* wic0 = P.Wic + (n0 + lm) * I_;
    const unsigned short* wic1 = P.Wic + (n1 + lm) * I_;
    const unsigned short* whi0 = P.Whi + (n0 + lm) * H_;
    const unsigned short* whi1 = P.Whi + (n1 + lm) * H_;
    const unsigned short* whf0 = P.Whf + (n0 + lm) * H_;
    const unsigned short* whf1 = P.Whf + (n1 + lm) * H_;
    const unsigned short* whc0 = P.Whc + (n0 + lm) * H_;
    const unsigned short* whc1 = P.Whc + (n1 + lm) * H_;
    const unsigned short* wcf0 = P.Wcf + (n0 + lm) * H_;
    const unsigned short* wcf1 = P.Wcf + (n1 + lm) * H_;

    v8f ai0 = {}, ai1 = {}, af0 = {}, af1 = {}, ac0 = {}, ac1 = {};

    for (int k = 0; k < I_; k += 32) {          // input projection, K = I
        v16bf a = load_a(xr, k, lane);
        ai0 = wmma_bf16(a, load_b(wii0, k, lane), ai0);
        ai1 = wmma_bf16(a, load_b(wii1, k, lane), ai1);
        af0 = wmma_bf16(a, load_b(wif0, k, lane), af0);
        af1 = wmma_bf16(a, load_b(wif1, k, lane), af1);
        ac0 = wmma_bf16(a, load_b(wic0, k, lane), ac0);
        ac1 = wmma_bf16(a, load_b(wic1, k, lane), ac1);
    }
    for (int k = 0; k < H_; k += 32) {          // recurrent projection, K = H
        v16bf a = load_a(hr, k, lane);
        ai0 = wmma_bf16(a, load_b(whi0, k, lane), ai0);
        ai1 = wmma_bf16(a, load_b(whi1, k, lane), ai1);
        af0 = wmma_bf16(a, load_b(whf0, k, lane), af0);
        af1 = wmma_bf16(a, load_b(whf1, k, lane), af1);
        ac0 = wmma_bf16(a, load_b(whc0, k, lane), ac0);
        ac1 = wmma_bf16(a, load_b(whc1, k, lane), ac1);
    }
    for (int k = 0; k < H_; k += 32) {          // f-gate peephole c @ Wcf^T
        v16bf a = load_a(cr, k, lane);
        af0 = wmma_bf16(a, load_b(wcf0, k, lane), af0);
        af1 = wmma_bf16(a, load_b(wcf1, k, lane), af1);
    }

    const int hi = lane >> 4;
    const float bi0 = P.bi[n0 + lm],  bi1 = P.bi[n1 + lm];
    const float bf0 = P.bfv[n0 + lm], bf1 = P.bfv[n1 + lm];
    const float bc0 = P.bc[n0 + lm],  bc1 = P.bc[n1 + lm];
#pragma unroll
    for (int r = 0; r < 8; r++) {
        int m = m0 + r + (hi << 3);
        size_t i0 = (size_t)m * H_ + n0 + lm;
        size_t i1 = (size_t)m * H_ + n1 + lm;

        float ig = sigm(ai0[r] + bi0);
        float fg = sigm(af0[r] + bf0);
        float g  = tanhf(ac0[r] + bc0);
        float cn = fg * P.c_f32[i0] + ig * g;
        P.c_f32[i0]  = cn;
        P.c_outb[i0] = f2bf(cn);

        ig = sigm(ai1[r] + bi1);
        fg = sigm(af1[r] + bf1);
        g  = tanhf(ac1[r] + bc1);
        cn = fg * P.c_f32[i1] + ig * g;
        P.c_f32[i1]  = cn;
        P.c_outb[i1] = f2bf(cn);
    }
}

// ---------------------------------------------------------------- step kernel B:
// zo = x_t@Wio^T + h@Who^T + c_new@Wco^T ; o = sigmoid(zo+bo); h_new = o*tanh(c_new)
struct OutParams {
    const unsigned short *x, *Wio, *Who, *Wco;
    const float* bo;
    const unsigned short *h_in, *c_newb;
    unsigned short* h_outb;
    const float* c_f32;
    float *h_f32, *o_f32;
    int t;
};

__global__ __launch_bounds__(256) void lstm_out_gate(OutParams P) {
    const int lane = threadIdx.x & 31;
    const int wave = threadIdx.x >> 5;
    const int lm   = lane & 15;
    const int m0   = blockIdx.y * 64 + (wave & 3) * 16;
    const int n0   = blockIdx.x * 64 + (wave >> 2) * 32;
    const int n1   = n0 + 16;

    const unsigned short* xr = P.x + (size_t)(m0 + lm) * (T_ * I_) + (size_t)P.t * I_;
    const unsigned short* hr = P.h_in + (m0 + lm) * H_;
    const unsigned short* cr = P.c_newb + (m0 + lm) * H_;

    const unsigned short* wio0 = P.Wio + (n0 + lm) * I_;
    const unsigned short* wio1 = P.Wio + (n1 + lm) * I_;
    const unsigned short* who0 = P.Who + (n0 + lm) * H_;
    const unsigned short* who1 = P.Who + (n1 + lm) * H_;
    const unsigned short* wco0 = P.Wco + (n0 + lm) * H_;
    const unsigned short* wco1 = P.Wco + (n1 + lm) * H_;

    v8f ao0 = {}, ao1 = {};

    for (int k = 0; k < I_; k += 32) {
        v16bf a = load_a(xr, k, lane);
        ao0 = wmma_bf16(a, load_b(wio0, k, lane), ao0);
        ao1 = wmma_bf16(a, load_b(wio1, k, lane), ao1);
    }
    for (int k = 0; k < H_; k += 32) {
        v16bf a = load_a(hr, k, lane);
        ao0 = wmma_bf16(a, load_b(who0, k, lane), ao0);
        ao1 = wmma_bf16(a, load_b(who1, k, lane), ao1);
    }
    for (int k = 0; k < H_; k += 32) {          // o-gate peephole c_new @ Wco^T
        v16bf a = load_a(cr, k, lane);
        ao0 = wmma_bf16(a, load_b(wco0, k, lane), ao0);
        ao1 = wmma_bf16(a, load_b(wco1, k, lane), ao1);
    }

    const int hi = lane >> 4;
    const float bo0 = P.bo[n0 + lm], bo1 = P.bo[n1 + lm];
#pragma unroll
    for (int r = 0; r < 8; r++) {
        int m = m0 + r + (hi << 3);
        size_t i0 = (size_t)m * H_ + n0 + lm;
        size_t i1 = (size_t)m * H_ + n1 + lm;

        float og = sigm(ao0[r] + bo0);
        float hn = og * tanhf(P.c_f32[i0]);
        P.o_f32[i0]  = og;
        P.h_f32[i0]  = hn;
        P.h_outb[i0] = f2bf(hn);

        og = sigm(ao1[r] + bo1);
        hn = og * tanhf(P.c_f32[i1]);
        P.o_f32[i1]  = og;
        P.h_f32[i1]  = hn;
        P.h_outb[i1] = f2bf(hn);
    }
}

// ---------------------------------------------------------------- host side
extern "C" void kernel_launch(void* const* d_in, const int* in_sizes, int n_in,
                              void* d_out, int out_size, void* d_ws, size_t ws_size,
                              hipStream_t stream) {
    const float* x   = (const float*)d_in[0];
    const float* Wii = (const float*)d_in[1];
    const float* Whi = (const float*)d_in[2];
    const float* bi  = (const float*)d_in[3];
    const float* Wif = (const float*)d_in[4];
    const float* Whf = (const float*)d_in[5];
    const float* Wcf = (const float*)d_in[6];
    const float* bf  = (const float*)d_in[7];
    const float* Wic = (const float*)d_in[8];
    const float* Whc = (const float*)d_in[9];
    const float* bc  = (const float*)d_in[10];
    const float* Wio = (const float*)d_in[11];
    const float* Who = (const float*)d_in[12];
    const float* Wco = (const float*)d_in[13];
    const float* bo  = (const float*)d_in[14];

    unsigned short* w = (unsigned short*)d_ws;
    size_t off = 0;
    auto alloc = [&](size_t elems) { unsigned short* p = w + off; off += elems; return p; };

    unsigned short* xb   = alloc((size_t)B_ * T_ * I_);
    unsigned short* WiiB = alloc((size_t)H_ * I_);
    unsigned short* WifB = alloc((size_t)H_ * I_);
    unsigned short* WicB = alloc((size_t)H_ * I_);
    unsigned short* WioB = alloc((size_t)H_ * I_);
    unsigned short* WhiB = alloc((size_t)H_ * H_);
    unsigned short* WhfB = alloc((size_t)H_ * H_);
    unsigned short* WhcB = alloc((size_t)H_ * H_);
    unsigned short* WhoB = alloc((size_t)H_ * H_);
    unsigned short* WcfB = alloc((size_t)H_ * H_);
    unsigned short* WcoB = alloc((size_t)H_ * H_);
    unsigned short* hb0  = alloc((size_t)B_ * H_);
    unsigned short* hb1  = alloc((size_t)B_ * H_);
    unsigned short* cb0  = alloc((size_t)B_ * H_);
    unsigned short* cb1  = alloc((size_t)B_ * H_);

    float* o_f32 = (float*)d_out;                       // outputs in return order (o, h, c)
    float* h_f32 = o_f32 + (size_t)B_ * H_;
    float* c_f32 = o_f32 + 2 * (size_t)B_ * H_;

    // bf16 conversions (weights stay L2-resident for the whole recurrence)
    cvt_f32_bf16<<<4096, 256, 0, stream>>>(x, xb, B_ * T_ * I_);
    cvt_f32_bf16<<<256, 256, 0, stream>>>(Wii, WiiB, H_ * I_);
    cvt_f32_bf16<<<256, 256, 0, stream>>>(Wif, WifB, H_ * I_);
    cvt_f32_bf16<<<256, 256, 0, stream>>>(Wic, WicB, H_ * I_);
    cvt_f32_bf16<<<256, 256, 0, stream>>>(Wio, WioB, H_ * I_);
    cvt_f32_bf16<<<512, 256, 0, stream>>>(Whi, WhiB, H_ * H_);
    cvt_f32_bf16<<<512, 256, 0, stream>>>(Whf, WhfB, H_ * H_);
    cvt_f32_bf16<<<512, 256, 0, stream>>>(Whc, WhcB, H_ * H_);
    cvt_f32_bf16<<<512, 256, 0, stream>>>(Who, WhoB, H_ * H_);
    cvt_f32_bf16<<<512, 256, 0, stream>>>(Wcf, WcfB, H_ * H_);
    cvt_f32_bf16<<<512, 256, 0, stream>>>(Wco, WcoB, H_ * H_);

    // zero initial state (h0 = c0 = 0); c_f32 lives in d_out and is RMW'd per step
    hipMemsetAsync(hb0, 0, (size_t)B_ * H_ * sizeof(unsigned short), stream);
    hipMemsetAsync(cb0, 0, (size_t)B_ * H_ * sizeof(unsigned short), stream);
    hipMemsetAsync(c_f32, 0, (size_t)B_ * H_ * sizeof(float), stream);

    dim3 grid(H_ / 64, B_ / 64);   // (8, 2) workgroups, 256 threads = 8 waves each
    for (int t = 0; t < T_; t++) {
        int p = t & 1;
        unsigned short* hin  = p ? hb1 : hb0;
        unsigned short* hout = p ? hb0 : hb1;
        unsigned short* cin  = p ? cb1 : cb0;
        unsigned short* cout = p ? cb0 : cb1;

        GateParams GP{ xb, WiiB, WhiB, WifB, WhfB, WcfB, WicB, WhcB,
                       bi, bf, bc, hin, cin, cout, c_f32, t };
        lstm_gates_ifc<<<grid, 256, 0, stream>>>(GP);

        OutParams OP{ xb, WioB, WhoB, WcoB, bo, hin, cout, hout,
                      c_f32, h_f32, o_f32, t };
        lstm_out_gate<<<grid, 256, 0, stream>>>(OP);
    }
}
